// FFTConv_18184891531616
// MI455X (gfx1250) — matrix operand order repaired and checked by reading, not compiled, not run on hardware
//
#include <hip/hip_runtime.h>

// FFT-free causal convolution y = conv_causal(h, x) + x*B as block-Toeplitz
// GEMM on v_wmma_f32_16x16x32_bf16.
//
//   Y_R = sum_k T_k X_{R-k},  T_k[i,j] = h[16k+i-j]
//
// Wave owns 8 adjacent output groups g0..g0+7 (each 256 timesteps).
// Toeplitz identity B(g,s) = B(g+1,s+8) lets ONE B load (x-operand, from L2,
// 128 B per WMMA) feed EIGHT WMMAs, each with a different A slice (h-Toeplitz)
// fetched from LDS:   acc_i += A_{s-56+8i} x B(g0+7, s)
// All groups end at the same step -> no wasted WMMAs. A is fetched as two
// aligned ds_load_b128 from 8 element-shifted LDS copies of reversed h
// (per-lane misalignment residue (-lane) mod 8 is constant per lane).

typedef __attribute__((ext_vector_type(16))) __bf16 v16bf;
typedef __attribute__((ext_vector_type(8)))  float  v8f;

#define D_DIM   1024
#define L_DIM   8192
#define BSZ     4
#define XB_PAD  272                        // front zeros absorb k>R boundary reads
#define XB_STRIDE (XB_PAD + L_DIM)         // 8464 elems; rows 16B-aligned
#define REV_LEN 8224                       // 17 front zeros + 8192 + 15 back zeros
#define NCOPY   8
#define LDS_BYTES (NCOPY * REV_LEN * 2)    // 131584 B (CDNA5 WGP: 320 KB)
#define NGRP    8                          // output groups per wave
#define PRO     (8 * (NGRP - 1))           // 56 prologue steps

static __device__ __forceinline__ unsigned short f32_to_bf16_rne(float f) {
  union { float f; unsigned int u; } v; v.f = f;
  unsigned int u = v.u;
  u += 0x7FFFu + ((u >> 16) & 1u);
  return (unsigned short)(u >> 16);
}

// x (fp32) -> zero-padded bf16 rows in workspace.
__global__ __launch_bounds__(256)
void xb_prep_kernel(const float* __restrict__ x, unsigned short* __restrict__ xb) {
  const int row = blockIdx.x;                       // b*D + d
  const float* src = x + (size_t)row * L_DIM;
  unsigned short* dst = xb + (size_t)row * XB_STRIDE;
  for (int i = threadIdx.x; i < XB_PAD; i += blockDim.x) dst[i] = 0;
  for (int t = threadIdx.x; t < L_DIM; t += blockDim.x)
    dst[XB_PAD + t] = f32_to_bf16_rne(src[t]);
}

// Workgroup = (d, oct-pair): 8 waves = 4 batches x 2 octs (oct = 8 groups).
__global__ __launch_bounds__(256)
void fftconv_wmma_kernel(const float* __restrict__ h,
                         const float* __restrict__ x,
                         const float* __restrict__ bias,
                         const unsigned short* __restrict__ xb,
                         float* __restrict__ out) {
  extern __shared__ unsigned short revc[];          // [NCOPY][REV_LEN]

  const int d  = blockIdx.x >> 1;
  const int op = blockIdx.x & 1;

  // revc[k][u] = REV[u+k], REV[w] = h[8208-w], zero for w<17 or w>8208.
  // (a) pad rows: only 32 of 8224 w's, no loads.
  for (int w = threadIdx.x; w < REV_LEN; w += 256) {
    if (w <= 16 || w >= 8209) {
#pragma unroll
      for (int k = 0; k < NCOPY; ++k)
        if (w >= k) revc[k * REV_LEN + (w - k)] = 0;
    }
  }
  // (b) fill: forward, coalesced, unconditional (w-k = 8208-t-k >= 10 always).
  const float* hrow = h + (size_t)d * L_DIM;
  for (int t = threadIdx.x; t < L_DIM; t += 256) {
    const unsigned short v = f32_to_bf16_rne(hrow[t]);
    const int w = 8208 - t;
#pragma unroll
    for (int k = 0; k < NCOPY; ++k)
      revc[k * REV_LEN + (w - k)] = v;
  }
  __syncthreads();

  const int wave = threadIdx.x >> 5;                // wave32
  const int lane = threadIdx.x & 31;
  const int lm = lane & 15;
  const int hi = lane >> 4;
  const int b  = wave & 3;
  const int oc = 2 * op + (wave >> 2);              // oct 0..3
  const int g0 = oc << 3;                           // groups g0..g0+7
  const int Send = 8 * g0 + 63;                     // common final step

  // A slice for group i at global step s: sigma_i = s - 56 + 8i,
  // slot-0 element = 8208 - 32*sigma - lm - 16hi = 10000 - 256i - 32s - lm - 16hi.
  const int shift = (-lm) & 7;                      // lane-constant residue
  const uint4* ap = (const uint4*)(revc + shift * REV_LEN);
  const int idx0 = (10000 - lm - 16 * hi - shift) >> 3;  // uint4 idx at s=0,i=0

  const unsigned short* xrow = xb + (size_t)(b * D_DIM + d) * XB_STRIDE;
  const int xb0 = XB_PAD + 16 * (16 * (g0 + NGRP - 1) + lm - hi); // B base, s=0

  v8f acc[NGRP];
#pragma unroll
  for (int i = 0; i < NGRP; ++i)
    acc[i] = (v8f){0.f, 0.f, 0.f, 0.f, 0.f, 0.f, 0.f, 0.f};

  auto step = [&](int s, int i, const v16bf& Bv_) {
    union { uint4 qq[2]; v16bf v; } A;
    const int bidx = idx0 - 4 * s - 32 * i;
    A.qq[0] = ap[bidx];                  // slots 0..7
    A.qq[1] = ap[bidx + 1];              // slots 8..15 (element base + 8)
    acc[i] = __builtin_amdgcn_wmma_f32_16x16x32_bf16(false, A.v, false, Bv_,
                                                     (short)0, acc[i],
                                                     false, false);
  };

  // Prologue: group i joins at s = 56-8i (uniform guards; EXEC stays full).
  for (int s = 0; s < PRO; ++s) {
    union { uint4 qq[2]; v16bf v; } Bm;
    const uint4* bp = (const uint4*)(xrow + (xb0 - 32 * s));
    Bm.qq[0] = bp[0];
    Bm.qq[1] = bp[1];
    step(s, 7, Bm.v);
    if (s >=  8) step(s, 6, Bm.v);
    if (s >= 16) step(s, 5, Bm.v);
    if (s >= 24) step(s, 4, Bm.v);
    if (s >= 32) step(s, 3, Bm.v);
    if (s >= 40) step(s, 2, Bm.v);
    if (s >= 48) step(s, 1, Bm.v);
  }

  // Main: one B load feeds eight WMMAs. Trip count = 8*g0+8 (multiple of 8).
#pragma unroll 2
  for (int s = PRO; s <= Send; ++s) {
    union { uint4 qq[2]; v16bf v; } Bm;
    const uint4* bp = (const uint4*)(xrow + (xb0 - 32 * s));
    Bm.qq[0] = bp[0];
    Bm.qq[1] = bp[1];
    if (s + 8 <= Send)                   // backward stride: prefetch explicitly
      __builtin_prefetch(xrow + (xb0 - 32 * (s + 8)), 0, 0);
#pragma unroll
    for (int i = 0; i < NGRP; ++i)
      step(s, i, Bm.v);
  }

  // Epilogue: acc[i][v] is t = 16*(16*(g0+i)+lm) + 8*hi + v; fuse x*B skip.
  const float bd = bias[d];
  const size_t rowoff = (size_t)(b * D_DIM + d) * L_DIM;
#pragma unroll
  for (int i = 0; i < NGRP; ++i) {
    const int t0 = 16 * (16 * (g0 + i) + lm) + 8 * hi;
    const size_t off = rowoff + t0;
    const float4* xs = (const float4*)(x + off);
    float4* dst = (float4*)(out + off);
    const float4 x0 = xs[0], x1 = xs[1];
    float4 o0, o1;
    o0.x = acc[i][0] + x0.x * bd; o0.y = acc[i][1] + x0.y * bd;
    o0.z = acc[i][2] + x0.z * bd; o0.w = acc[i][3] + x0.w * bd;
    o1.x = acc[i][4] + x1.x * bd; o1.y = acc[i][5] + x1.y * bd;
    o1.z = acc[i][6] + x1.z * bd; o1.w = acc[i][7] + x1.w * bd;
    dst[0] = o0;
    dst[1] = o1;
  }
}

extern "C" void kernel_launch(void* const* d_in, const int* in_sizes, int n_in,
                              void* d_out, int out_size, void* d_ws, size_t ws_size,
                              hipStream_t stream) {
  const float* h  = (const float*)d_in[0];    // [D, L] fp32
  const float* x  = (const float*)d_in[1];    // [Bsz, D, L] fp32
  const float* Bv = (const float*)d_in[2];    // [D, 1] fp32
  float* out = (float*)d_out;                 // [Bsz, D, L] fp32
  unsigned short* xb = (unsigned short*)d_ws; // needs BSZ*D*XB_STRIDE*2 ~ 69.3 MB

  xb_prep_kernel<<<BSZ * D_DIM, 256, 0, stream>>>(x, xb);
  fftconv_wmma_kernel<<<D_DIM * 2, 256, LDS_BYTES, stream>>>(h, x, Bv, xb, out);
}
